// SelfAttention_42399917146860
// MI455X (gfx1250) — matrix operand order repaired and checked by reading, not compiled, not run on hardware
//
#include <hip/hip_runtime.h>
#include <hip/hip_bf16.h>

// Problem constants (fixed by the reference)
#define S_LEN  2048
#define EMB    1024
#define NH     16
#define DH     64
#define BATCH  2
#define M_ROWS (BATCH * S_LEN)   // 4096

typedef __attribute__((ext_vector_type(16))) __bf16 v16bf;
typedef __attribute__((ext_vector_type(8)))  __bf16 v8bf;
typedef __attribute__((ext_vector_type(8)))  float  v8f;
typedef __attribute__((ext_vector_type(4)))  unsigned int u32x4;
typedef __attribute__((ext_vector_type(8)))  int         i32x8;
typedef __attribute__((ext_vector_type(4)))  int         i32x4;

// Detect the TDM builtins (tensor data mover). Fallback: cooperative ds stores.
#if defined(__has_builtin)
#if __has_builtin(__builtin_amdgcn_tensor_load_to_lds) && \
    __has_builtin(__builtin_amdgcn_s_wait_tensorcnt)
#define HAVE_TDM 1
#endif
#endif

// ---------------------------------------------------------------------------
// fp32 -> bf16 (round to nearest even)
// ---------------------------------------------------------------------------
__device__ __forceinline__ unsigned short f2bf(float f) {
    unsigned int u = __float_as_uint(f);
    u += 0x7fffu + ((u >> 16) & 1u);
    return (unsigned short)(u >> 16);
}

// ---------------------------------------------------------------------------
// DPP16 lane permute (VALU, no LDS): returns x from the DPP-selected lane.
// Reduction over each 16-lane half-row: quad xor1, quad xor2,
// row_half_mirror (0x141), row_mirror (0x140).
// ---------------------------------------------------------------------------
template<int CTRL>
__device__ __forceinline__ float dpp_movf(float x) {
    return __int_as_float(__builtin_amdgcn_update_dpp(
        0, __float_as_int(x), CTRL, 0xF, 0xF, true));
}

__device__ __forceinline__ void row_reduce_max8(float (&v)[8]) {
#pragma unroll
    for (int r = 0; r < 8; r++) v[r] = fmaxf(v[r], dpp_movf<0xB1>(v[r]));  // xor1
#pragma unroll
    for (int r = 0; r < 8; r++) v[r] = fmaxf(v[r], dpp_movf<0x4E>(v[r]));  // xor2
#pragma unroll
    for (int r = 0; r < 8; r++) v[r] = fmaxf(v[r], dpp_movf<0x141>(v[r])); // half_mirror
#pragma unroll
    for (int r = 0; r < 8; r++) v[r] = fmaxf(v[r], dpp_movf<0x140>(v[r])); // mirror
}

__device__ __forceinline__ void row_reduce_sum8(float (&v)[8]) {
#pragma unroll
    for (int r = 0; r < 8; r++) v[r] += dpp_movf<0xB1>(v[r]);
#pragma unroll
    for (int r = 0; r < 8; r++) v[r] += dpp_movf<0x4E>(v[r]);
#pragma unroll
    for (int r = 0; r < 8; r++) v[r] += dpp_movf<0x141>(v[r]);
#pragma unroll
    for (int r = 0; r < 8; r++) v[r] += dpp_movf<0x140>(v[r]);
}

// ---------------------------------------------------------------------------
// WMMA fragment loaders (CDNA5 wave32 16-bit layouts, CDNA5 ISA)
// ---------------------------------------------------------------------------
__device__ __forceinline__ v16bf load_a_frag(const unsigned short* base, int ld, int lane) {
    const int m  = lane & 15;
    const int kh = (lane >> 4) & 1;
    v8bf lo = *(const v8bf*)(base + (size_t)m * ld + kh * 8);
    v8bf hi = *(const v8bf*)(base + (size_t)m * ld + 16 + kh * 8);
    return __builtin_shufflevector(lo, hi, 0,1,2,3,4,5,6,7,8,9,10,11,12,13,14,15);
}

__device__ __forceinline__ v16bf load_b_frag(const unsigned short* base, int ld, int lane) {
    const int n  = lane & 15;
    const int kh = (lane >> 4) & 1;
    v8bf lo = *(const v8bf*)(base + (size_t)n * ld + kh * 8);
    v8bf hi = *(const v8bf*)(base + (size_t)n * ld + 16 + kh * 8);
    return __builtin_shufflevector(lo, hi, 0,1,2,3,4,5,6,7,8,9,10,11,12,13,14,15);
}

__device__ __forceinline__ v8f wmma_bf16(v16bf a, v16bf b, v8f c) {
    return __builtin_amdgcn_wmma_f32_16x16x32_bf16(
        false, a, false, b, (short)0, c, false, false);
}

// ---------------------------------------------------------------------------
// TDM: async DMA of a [rows x DH] bf16 tile (row-major, row stride DH elems)
// from global memory into LDS at byte offset lds_off.
// Descriptor bit packing per CDNA5 ISA.
// ---------------------------------------------------------------------------
#if HAVE_TDM
__device__ __forceinline__ void tdm_load_tile(unsigned lds_off,
                                              const unsigned short* gptr,
                                              int rows) {
    unsigned long long ga = (unsigned long long)(size_t)gptr;
    u32x4 g0;
    g0[0] = 1u;                                   // count=1, user descriptor
    g0[1] = lds_off;                              // lds_addr (bytes)
    g0[2] = (unsigned)ga;                         // global_addr[31:0]
    g0[3] = (unsigned)((ga >> 32) & 0x1FFFFFFu)   // global_addr[56:32]
          | (2u << 30);                           // type=2 ("image")
    i32x8 g1;
    g1[0] = (int)(1u << 16);          // workgroup_mask=0, data_size=1 (2B)
    g1[1] = (int)((unsigned)DH << 16);        // tensor_dim0 low16 = 64
    g1[2] = (int)((unsigned)S_LEN << 16);     // tensor_dim0 hi=0 | tensor_dim1 lo16
    g1[3] = (int)((unsigned)DH << 16);        // tensor_dim1 hi=0 | tile_dim0 = 64
    g1[4] = rows;                             // tile_dim1 = rows, tile_dim2 = 0
    g1[5] = DH;                               // tensor_dim0_stride low32 = 64
    g1[6] = 0;                                // stride hi | tensor_dim1_stride lo
    g1[7] = 0;                                // tensor_dim1_stride hi
    i32x4 z4 = {0, 0, 0, 0};
#if defined(__clang_major__) && (__clang_major__ >= 23)
    i32x8 z8 = {};
    __builtin_amdgcn_tensor_load_to_lds(g0, g1, z4, z4, z8, 0);
#else
    __builtin_amdgcn_tensor_load_to_lds(g0, g1, z4, z4, 0);
#endif
}
#endif

// ---------------------------------------------------------------------------
// fp32 -> bf16 conversion kernel
// ---------------------------------------------------------------------------
__global__ void cvt_f32_bf16_kernel(const float* __restrict__ in,
                                    unsigned short* __restrict__ out, int n) {
    int i = blockIdx.x * 256 + threadIdx.x;
    if (i < n) out[i] = f2bf(in[i]);
}

// ---------------------------------------------------------------------------
// Tiled WMMA GEMM: C[m,n] = (sum_k A[m,k]*W[n,k] + bias[n]) * scale
//   MODE 1: bf16 out, head layout [B,NH,S,DH] (Q/K/V)
//   MODE 2: f32 out, row-major [M_ROWS, EMB]  (final projection)
// Block = 128 threads (4 waves); block tile 64x128; wave tile 32x64
// ---------------------------------------------------------------------------
template<int MODE>
__global__ __launch_bounds__(128) void gemm_bf16_kernel(
    const unsigned short* __restrict__ A,
    const unsigned short* __restrict__ Bw,
    const float* __restrict__ bias,
    unsigned short* __restrict__ outH,
    float* __restrict__ outF,
    int K, float scale)
{
    const int lane = threadIdx.x & 31;
    const int wave = threadIdx.x >> 5;
    const int m0 = blockIdx.y * 64 + (wave >> 1) * 32;
    const int n0 = blockIdx.x * 128 + (wave & 1) * 64;

    v8f acc0[4] = {v8f{}, v8f{}, v8f{}, v8f{}};
    v8f acc1[4] = {v8f{}, v8f{}, v8f{}, v8f{}};

    const unsigned short* a0p = A + (size_t)m0 * K;
    const unsigned short* a1p = A + (size_t)(m0 + 16) * K;
    const unsigned short* bp  = Bw + (size_t)n0 * K;

    for (int k0 = 0; k0 < K; k0 += 32) {
        __builtin_prefetch(a0p + k0 + 256, 0, 3);   // global_prefetch_b8
        __builtin_prefetch(bp  + k0 + 256, 0, 3);
        v16bf a0 = load_a_frag(a0p + k0, K, lane);
        v16bf a1 = load_a_frag(a1p + k0, K, lane);
        v16bf b[4];
#pragma unroll
        for (int j = 0; j < 4; j++)
            b[j] = load_b_frag(bp + (size_t)(16 * j) * K + k0, K, lane);
#pragma unroll
        for (int j = 0; j < 4; j++) {
            acc0[j] = wmma_bf16(a0, b[j], acc0[j]);
            acc1[j] = wmma_bf16(a1, b[j], acc1[j]);
        }
    }

    // C layout: lane l, VGPR r -> row = r + 8*(l>>4), col = l&15
    const int rowHi = 8 * (lane >> 4);
    const int colLo = lane & 15;
#pragma unroll
    for (int j = 0; j < 4; j++) {
        const int col = n0 + j * 16 + colLo;
        const float bcol = bias[col];
        const size_t colOff = (MODE == 1)
            ? ((size_t)(col >> 6) * (S_LEN * DH) + (size_t)(col & (DH - 1)))
            : (size_t)col;
#pragma unroll
        for (int i = 0; i < 2; i++) {
            v8f acc = (i == 0) ? acc0[j] : acc1[j];
            const int rbase = m0 + i * 16 + rowHi;
#pragma unroll
            for (int r = 0; r < 8; r++) {
                const int row = rbase + r;
                const float v = (acc[r] + bcol) * scale;
                if (MODE == 1) {
                    const int b2 = row >> 11;             // batch
                    const int s  = row & (S_LEN - 1);     // sequence pos
                    outH[(size_t)b2 * (NH * S_LEN * DH) +
                         (size_t)s * DH + colOff] = f2bf(v);
                } else {
                    outF[(size_t)row * EMB + colOff] = v;
                }
            }
        }
    }
}

// ---------------------------------------------------------------------------
// Flash attention: one block = 64 query rows of one (b,h); 4 waves x 16 rows.
// K tiles streamed by the Tensor Data Mover (double-buffered); V transposed
// into LDS by the waves. WMMA matmuls; online softmax in f32, base-2 domain
// (log2(e)/sqrt(DH) folded into Q), DPP16 VALU row reductions (no LDS).
// ---------------------------------------------------------------------------
__global__ __launch_bounds__(128) void attn_kernel(
    const unsigned short* __restrict__ Qg,
    const unsigned short* __restrict__ Kg,
    const unsigned short* __restrict__ Vg,
    unsigned short* __restrict__ Og)
{
    __shared__ __align__(32) unsigned short KtBuf[2][32 * DH]; // 2 x 4 KB
    __shared__ __align__(32) unsigned short Vt[DH * 32];       // [d][key] 4 KB
    __shared__ __align__(32) unsigned short Pt[4][16 * 32];    // per-wave P

    const int tid  = threadIdx.x;
    const int lane = tid & 31;
    const int wave = tid >> 5;
    const int bh   = blockIdx.x >> 5;   // (b*NH + h)
    const int qblk = blockIdx.x & 31;
    const int q0   = qblk * 64 + wave * 16;

    const unsigned short* Qb = Qg + (size_t)bh * S_LEN * DH;
    const unsigned short* Kb = Kg + (size_t)bh * S_LEN * DH;
    const unsigned short* Vb = Vg + (size_t)bh * S_LEN * DH;

    v16bf qa0 = load_a_frag(Qb + (size_t)q0 * DH,      DH, lane);
    v16bf qa1 = load_a_frag(Qb + (size_t)q0 * DH + 32, DH, lane);

    v8f o0 = {}, o1 = {}, o2 = {}, o3 = {};
    float mrow[8], lrow[8];
#pragma unroll
    for (int r = 0; r < 8; r++) { mrow[r] = -3.0e38f; lrow[r] = 0.0f; }

    unsigned short* P = &Pt[wave][0];
    const int rowHi = 8 * (lane >> 4);
    const int colLo = lane & 15;

#if HAVE_TDM
    // Prime the pipeline: DMA key block 0 into buffer 0.
    if (wave == 0)
        tdm_load_tile((unsigned)(size_t)(void*)&KtBuf[0][0], Kb, 32);
#endif

    for (int j0 = 0; j0 < S_LEN; j0 += 32) {
        const int cur = (j0 >> 5) & 1;
        __syncthreads();   // everyone done reading previous LDS contents

#if HAVE_TDM
        // Kick off DMA of the NEXT key block into the other buffer.
        if (wave == 0 && (j0 + 32) < S_LEN)
            tdm_load_tile((unsigned)(size_t)(void*)&KtBuf[cur ^ 1][0],
                          Kb + (size_t)(j0 + 32) * DH, 32);
#endif
        // Stage V transposed (d-major) so P@V B-fragments are contiguous.
#pragma unroll 4
        for (int t = tid; t < 32 * DH; t += 128) {
            int key = t >> 6, d = t & (DH - 1);
            Vt[d * 32 + key] = Vb[(size_t)(j0 + key) * DH + d];
#if !HAVE_TDM
            KtBuf[cur][key * DH + d] = Kb[(size_t)(j0 + key) * DH + d];
#endif
        }
        if (j0 + 32 < S_LEN)
            __builtin_prefetch(Vb + (size_t)(j0 + 32) * DH + lane * 8, 0, 0);

#if HAVE_TDM
        if (wave == 0) {
            if ((j0 + 32) < S_LEN) __builtin_amdgcn_s_wait_tensorcnt(1);
            else                   __builtin_amdgcn_s_wait_tensorcnt(0);
        }
#endif
        __syncthreads();   // Kt[cur] (DMA) + Vt (stores) visible to all waves

        const unsigned short* Kt = &KtBuf[cur][0];

        // S = Q @ K^T : 16 queries x 32 keys, contraction over D=64
        v8f s0 = {}, s1 = {};
        s0 = wmma_bf16(qa0, load_b_frag(Kt,                DH, lane), s0);
        s0 = wmma_bf16(qa1, load_b_frag(Kt + 32,           DH, lane), s0);
        s1 = wmma_bf16(qa0, load_b_frag(Kt + 16 * DH,      DH, lane), s1);
        s1 = wmma_bf16(qa1, load_b_frag(Kt + 16 * DH + 32, DH, lane), s1);

        // --- online softmax (base-2 domain), DPP16 VALU reductions ---
        float mx[8];
#pragma unroll
        for (int r = 0; r < 8; r++) mx[r] = fmaxf(s0[r], s1[r]);
        row_reduce_max8(mx);

        float alpha[8];
#pragma unroll
        for (int r = 0; r < 8; r++) {
            float mn = fmaxf(mrow[r], mx[r]);
            alpha[r] = exp2f(mrow[r] - mn);
            mrow[r] = mn;
        }
        float rs[8];
#pragma unroll
        for (int r = 0; r < 8; r++) {
            s0[r] = exp2f(s0[r] - mrow[r]);
            s1[r] = exp2f(s1[r] - mrow[r]);
            rs[r] = s0[r] + s1[r];
        }
        row_reduce_sum8(rs);
#pragma unroll
        for (int r = 0; r < 8; r++) {
            lrow[r] = lrow[r] * alpha[r] + rs[r];
            o0[r] *= alpha[r]; o1[r] *= alpha[r];
            o2[r] *= alpha[r]; o3[r] *= alpha[r];
        }

        // P (C layout) -> LDS row-major [16 q][32 key] -> reload as A fragment
#pragma unroll
        for (int r = 0; r < 8; r++) {
            P[(rowHi + r) * 32 + colLo]      = f2bf(s0[r]);
            P[(rowHi + r) * 32 + 16 + colLo] = f2bf(s1[r]);
        }
        v16bf pa = load_a_frag(P, 32, lane);

        // O += P @ V
        o0 = wmma_bf16(pa, load_b_frag(Vt,           32, lane), o0);
        o1 = wmma_bf16(pa, load_b_frag(Vt + 16 * 32, 32, lane), o1);
        o2 = wmma_bf16(pa, load_b_frag(Vt + 32 * 32, 32, lane), o2);
        o3 = wmma_bf16(pa, load_b_frag(Vt + 48 * 32, 32, lane), o3);
    }

    // Normalize and write O as [B, S, EMB] bf16 (feeds final projection GEMM)
    const int b = bh >> 4, h = bh & 15;
#pragma unroll
    for (int r = 0; r < 8; r++) {
        float inv = 1.0f / lrow[r];
        int srow = q0 + rowHi + r;
        size_t base = ((size_t)(b * S_LEN + srow)) * EMB + h * DH;
        Og[base +  0 + colLo] = f2bf(o0[r] * inv);
        Og[base + 16 + colLo] = f2bf(o1[r] * inv);
        Og[base + 32 + colLo] = f2bf(o2[r] * inv);
        Og[base + 48 + colLo] = f2bf(o3[r] * inv);
    }
}

// ---------------------------------------------------------------------------
// Host-side orchestration
// ---------------------------------------------------------------------------
extern "C" void kernel_launch(void* const* d_in, const int* in_sizes, int n_in,
                              void* d_out, int out_size, void* d_ws, size_t ws_size,
                              hipStream_t stream)
{
    (void)in_sizes; (void)n_in; (void)out_size; (void)ws_size;

    const float* x  = (const float*)d_in[0];
    const float* Wq = (const float*)d_in[1];
    const float* bq = (const float*)d_in[2];
    const float* Wk = (const float*)d_in[3];
    const float* bk = (const float*)d_in[4];
    const float* Wv = (const float*)d_in[5];
    const float* bv = (const float*)d_in[6];
    const float* Wo = (const float*)d_in[7];
    const float* bo = (const float*)d_in[8];
    float* out = (float*)d_out;

    // Workspace partition (bf16 buffers), total ~48 MiB
    char* ws = (char*)d_ws;
    unsigned short* xb  = (unsigned short*)ws; ws += (size_t)M_ROWS * EMB * 2;
    unsigned short* Wqb = (unsigned short*)ws; ws += (size_t)EMB * EMB * 2;
    unsigned short* Wkb = (unsigned short*)ws; ws += (size_t)EMB * EMB * 2;
    unsigned short* Wvb = (unsigned short*)ws; ws += (size_t)EMB * EMB * 2;
    unsigned short* Wob = (unsigned short*)ws; ws += (size_t)EMB * EMB * 2;
    unsigned short* Qb  = (unsigned short*)ws; ws += (size_t)M_ROWS * EMB * 2;
    unsigned short* Kb  = (unsigned short*)ws; ws += (size_t)M_ROWS * EMB * 2;
    unsigned short* Vb  = (unsigned short*)ws; ws += (size_t)M_ROWS * EMB * 2;
    unsigned short* Ob  = (unsigned short*)ws; ws += (size_t)M_ROWS * EMB * 2;

    // 1) fp32 -> bf16 conversions
    const int nx = M_ROWS * EMB;
    const int nw = EMB * EMB;
    cvt_f32_bf16_kernel<<<(nx + 255) / 256, 256, 0, stream>>>(x,  xb,  nx);
    cvt_f32_bf16_kernel<<<(nw + 255) / 256, 256, 0, stream>>>(Wq, Wqb, nw);
    cvt_f32_bf16_kernel<<<(nw + 255) / 256, 256, 0, stream>>>(Wk, Wkb, nw);
    cvt_f32_bf16_kernel<<<(nw + 255) / 256, 256, 0, stream>>>(Wv, Wvb, nw);
    cvt_f32_bf16_kernel<<<(nw + 255) / 256, 256, 0, stream>>>(Wo, Wob, nw);

    // 2) QKV projections -> head-major bf16.
    //    Q is pre-scaled by log2(e)/sqrt(DH) so softmax runs in base-2
    //    (exp2f only, no per-element multiplies by log2e).
    dim3 gemm_grid(EMB / 128, M_ROWS / 64);  // (8, 64)
    const float qscale = 0.125f * 1.44269504088896340736f;
    gemm_bf16_kernel<1><<<gemm_grid, 128, 0, stream>>>(xb, Wqb, bq, Qb, nullptr,
                                                       EMB, qscale);
    gemm_bf16_kernel<1><<<gemm_grid, 128, 0, stream>>>(xb, Wkb, bk, Kb, nullptr,
                                                       EMB, 1.0f);
    gemm_bf16_kernel<1><<<gemm_grid, 128, 0, stream>>>(xb, Wvb, bv, Vb, nullptr,
                                                       EMB, 1.0f);

    // 3) Flash attention: B*NH*(S/64) = 1024 blocks of 128 threads
    attn_kernel<<<dim3(BATCH * NH * (S_LEN / 64)), 128, 0, stream>>>(Qb, Kb, Vb, Ob);

    // 4) Output projection -> fp32 result
    gemm_bf16_kernel<2><<<gemm_grid, 128, 0, stream>>>(Ob, Wob, bo, nullptr, out,
                                                       EMB, 1.0f);
}
